// DeepSetLayer_87110526697907
// MI455X (gfx1250) — matrix operand order repaired and checked by reading, not compiled, not run on hardware
//
#include <hip/hip_runtime.h>
#include <hip/hip_bf16.h>

// ---------------------------------------------------------------------------
// DeepSet / GNN mean-aggregation layer for MI455X (gfx1250, wave32, WMMA)
//   msg_sum[d] += x[s] over edges; mean = msg_sum / max(counts,1)
//   out = x@W1 + b1 + (x - mean)@W2 + b2 ; rows with no messages pass x through
// Pipeline: zero -> edge scatter (f32 atomics) -> bf16 prep -> WMMA dual-GEMM.
// ---------------------------------------------------------------------------

#define D_FEAT 512

typedef __bf16  v16bf __attribute__((ext_vector_type(16)));
typedef float   v8f   __attribute__((ext_vector_type(8)));

__device__ __forceinline__ unsigned short f2bf(float f) {
    unsigned int a = __float_as_uint(f);
    a = a + 0x7FFFu + ((a >> 16) & 1u);       // round-to-nearest-even
    return (unsigned short)(a >> 16);
}

__device__ __forceinline__ unsigned int pack_bf16x2(float lo, float hi) {
    return (unsigned int)f2bf(lo) | ((unsigned int)f2bf(hi) << 16);
}

union FragAB {
    unsigned short s[16];
    unsigned int   u[8];
    uint4          q[2];
    v16bf          v;
};

// ---------------------------------------------------------------------------
// Zero workspace (msg_sum + counts)
// ---------------------------------------------------------------------------
__global__ void zero_kernel(float* __restrict__ p, long long n) {
    long long i = (long long)blockIdx.x * blockDim.x + threadIdx.x;
    if (i < n) p[i] = 0.0f;
}

// ---------------------------------------------------------------------------
// Edge scatter: one block per edge, 128 lanes x float4 (512 floats per row).
// ---------------------------------------------------------------------------
__global__ __launch_bounds__(128)
void scatter_kernel(const float* __restrict__ x,
                    const int*   __restrict__ src,
                    const int*   __restrict__ dst,
                    float*       __restrict__ msg,
                    float*       __restrict__ counts,
                    int n_edges) {
    int e = blockIdx.x;
    if (e >= n_edges) return;
    int s = src[e];
    int d = dst[e];
    const float4* xr = (const float4*)(x + (size_t)s * D_FEAT);
    float*        mr = msg + (size_t)d * D_FEAT;

    float4 v = xr[threadIdx.x];
    int f = threadIdx.x * 4;
    atomicAdd(mr + f + 0, v.x);
    atomicAdd(mr + f + 1, v.y);
    atomicAdd(mr + f + 2, v.z);
    atomicAdd(mr + f + 3, v.w);
    if (threadIdx.x == 0) atomicAdd(counts + d, 1.0f);
}

// ---------------------------------------------------------------------------
// Prep: xbf = bf16(x); xmbf = bf16(x - msg/max(counts,1)).
// One thread per 8 consecutive elements (same row since 512 % 8 == 0):
// 2x float4 loads from x and msg, 1x b128 store to each bf16 array.
// ---------------------------------------------------------------------------
__global__ __launch_bounds__(256)
void prep_kernel(const float* __restrict__ x,
                 const float* __restrict__ msg,
                 const float* __restrict__ counts,
                 unsigned short* __restrict__ xbf,
                 unsigned short* __restrict__ xmbf,
                 long long n_elems) {
    long long i0 = ((long long)blockIdx.x * blockDim.x + threadIdx.x) * 8;
    if (i0 >= n_elems) return;
    long long row = i0 >> 9;                 // / D_FEAT
    float invc = 1.0f / fmaxf(counts[row], 1.0f);

    float4 x0 = *(const float4*)(x + i0);
    float4 x1 = *(const float4*)(x + i0 + 4);
    float4 m0 = *(const float4*)(msg + i0);
    float4 m1 = *(const float4*)(msg + i0 + 4);

    uint4 xo, mo;
    xo.x = pack_bf16x2(x0.x, x0.y);
    xo.y = pack_bf16x2(x0.z, x0.w);
    xo.z = pack_bf16x2(x1.x, x1.y);
    xo.w = pack_bf16x2(x1.z, x1.w);
    mo.x = pack_bf16x2(x0.x - m0.x * invc, x0.y - m0.y * invc);
    mo.y = pack_bf16x2(x0.z - m0.z * invc, x0.w - m0.w * invc);
    mo.z = pack_bf16x2(x1.x - m1.x * invc, x1.y - m1.y * invc);
    mo.w = pack_bf16x2(x1.z - m1.z * invc, x1.w - m1.w * invc);

    *(uint4*)(xbf + i0)  = xo;
    *(uint4*)(xmbf + i0) = mo;
}

// ---------------------------------------------------------------------------
// Fused dual-GEMM: out = x@W1 + (x-mean)@W2 + (b1+b2), passthrough counts==0.
// Block: 256 threads = 8 waves; block tile = 128 rows x 64 cols.
// Each wave: 16 rows x 64 cols = 4 accumulators; 8 WMMAs per K-step of 32.
// W1/W2 panels staged once in LDS, pre-swizzled so each B fragment is two
// contiguous ds_load_b128 per lane.
// ---------------------------------------------------------------------------
__global__ __launch_bounds__(256)
void fused_gemm_kernel(const unsigned short* __restrict__ xbf,
                       const unsigned short* __restrict__ xmbf,
                       const float* __restrict__ x,
                       const float* __restrict__ counts,
                       const float* __restrict__ W1,
                       const float* __restrict__ b1,
                       const float* __restrict__ W2,
                       const float* __restrict__ b2,
                       float*       __restrict__ out,
                       int n_nodes) {
    // [mat(2)][coltile(4)][kt(16)][lane(32)][p(8)] packed bf16x2 dwords = 128KB
    __shared__ unsigned int lw[2 * 4 * 16 * 32 * 8];

    const int n0  = blockIdx.x * 64;     // column origin of block
    const int tid = threadIdx.x;

    // ---- stage W panels, pre-swizzled to per-lane fragment order
    for (int i = tid; i < 2 * 4 * 4096; i += 256) {
        int mat  = i >> 14;
        int rem  = i & 16383;
        int ct   = rem >> 12;
        int idx  = rem & 4095;
        int kt   = idx >> 8;
        int lane = (idx >> 3) & 31;
        int p    = idx & 7;
        int n    = lane & 15;
        int hi   = lane >> 4;
        int k2   = kt * 16 + hi * 4 + p + (p & 4); // K-pair (K = 2*k2, 2*k2+1)
        int col  = n0 + ct * 16 + n;
        const float* W = mat ? W2 : W1;
        lw[i] = pack_bf16x2(W[(2 * k2) * D_FEAT + col],
                            W[(2 * k2 + 1) * D_FEAT + col]);
    }
    __syncthreads();

    const int wave = tid >> 5;
    const int lane = tid & 31;
    const int hi   = lane >> 4;          // 0: lanes 0-15, 1: lanes 16-31
    const int bn   = lane & 15;

    const int m0     = blockIdx.y * 128 + wave * 16;
    const int arow   = m0 + bn;
    const int arow_c = (arow < n_nodes) ? arow : (n_nodes - 1);
    const int khalf  = hi << 3;          // lanes 16-31 own K offset +8

    const unsigned short* xrow = xbf  + (size_t)arow_c * D_FEAT;
    const unsigned short* mrow = xmbf + (size_t)arow_c * D_FEAT;

    v8f acc0 = {}, acc1 = {}, acc2 = {}, acc3 = {};

    #pragma unroll 4
    for (int kt = 0; kt < 16; ++kt) {
        const int kA = kt * 32 + khalf;

        if (kt < 15) {
            __builtin_prefetch(xrow + kA + 32, 0, 3);   // global_prefetch_b8
            __builtin_prefetch(mrow + kA + 32, 0, 3);
        }

        // A fragments (16x32 bf16, ISA layout): K kA..kA+7, kA+16..kA+23
        FragAB ax, am;
        ax.q[0] = *(const uint4*)(xrow + kA);
        ax.q[1] = *(const uint4*)(xrow + kA + 16);
        am.q[0] = *(const uint4*)(mrow + kA);
        am.q[1] = *(const uint4*)(mrow + kA + 16);

        const int fbase = (kt << 8) + (lane << 3);

        #pragma unroll
        for (int ct = 0; ct < 4; ++ct) {
            const unsigned int* p1 = &lw[(ct << 12) + fbase];
            const unsigned int* p2 = &lw[(1 << 14) + (ct << 12) + fbase];
            FragAB bw1, bw2;
            bw1.q[0] = *(const uint4*)(p1);
            bw1.q[1] = *(const uint4*)(p1 + 4);
            bw2.q[0] = *(const uint4*)(p2);
            bw2.q[1] = *(const uint4*)(p2 + 4);

            v8f a = (ct == 0) ? acc0 : (ct == 1) ? acc1 : (ct == 2) ? acc2 : acc3;
            a = __builtin_amdgcn_wmma_f32_16x16x32_bf16(
                    false, ax.v, false, bw1.v, (short)0, a, false, false);
            a = __builtin_amdgcn_wmma_f32_16x16x32_bf16(
                    false, am.v, false, bw2.v, (short)0, a, false, false);
            if (ct == 0) acc0 = a; else if (ct == 1) acc1 = a;
            else if (ct == 2) acc2 = a; else acc3 = a;
        }
    }

    // ---- epilogue (C layout: VGPR j -> row m0 + j + 8*hi, col per tile)
    const int rbase = m0 + (hi << 3);
    float bias[4];
    #pragma unroll
    for (int ct = 0; ct < 4; ++ct) {
        int col = n0 + ct * 16 + bn;
        bias[ct] = b1[col] + b2[col];
    }
    #pragma unroll
    for (int j = 0; j < 8; ++j) {
        int r = rbase + j;
        if (r < n_nodes) {
            bool has_msg = counts[r] > 0.0f;
            #pragma unroll
            for (int ct = 0; ct < 4; ++ct) {
                int col = n0 + ct * 16 + bn;
                float a = (ct == 0) ? acc0[j] : (ct == 1) ? acc1[j]
                        : (ct == 2) ? acc2[j] : acc3[j];
                float v = has_msg ? (a + bias[ct])
                                  : x[(size_t)r * D_FEAT + col];
                out[(size_t)r * D_FEAT + col] = v;
            }
        }
    }
}

// ---------------------------------------------------------------------------
// Launch
// ---------------------------------------------------------------------------
extern "C" void kernel_launch(void* const* d_in, const int* in_sizes, int n_in,
                              void* d_out, int out_size, void* d_ws, size_t ws_size,
                              hipStream_t stream) {
    const float* x   = (const float*)d_in[0];
    const int*   src = (const int*)  d_in[1];
    const int*   dst = (const int*)  d_in[2];
    const float* W1  = (const float*)d_in[3];
    const float* b1  = (const float*)d_in[4];
    const float* W2  = (const float*)d_in[5];
    const float* b2  = (const float*)d_in[6];
    float*       out = (float*)d_out;

    const int n_nodes = in_sizes[0] / D_FEAT;
    const int n_edges = in_sizes[1];
    const long long n_elems = (long long)n_nodes * D_FEAT;

    // workspace layout: msg f32 | counts f32 | xbf bf16 | xmbf bf16
    char* ws = (char*)d_ws;
    float* msg    = (float*)ws;                               ws += n_elems * 4;
    float* counts = (float*)ws;                               ws += (size_t)n_nodes * 4;
    ws = (char*)(((uintptr_t)ws + 15) & ~(uintptr_t)15);
    unsigned short* xbf  = (unsigned short*)ws;               ws += n_elems * 2;
    unsigned short* xmbf = (unsigned short*)ws;

    // 1) zero msg_sum + counts
    long long nz = n_elems + n_nodes;
    zero_kernel<<<(int)((nz + 255) / 256), 256, 0, stream>>>(msg, nz);

    // 2) edge scatter (gather x[src], atomic-add into msg[dst], count edges)
    scatter_kernel<<<n_edges, 128, 0, stream>>>(x, src, dst, msg, counts, n_edges);

    // 3) bf16 prep of x and (x - mean)
    long long nthreads = n_elems / 8;
    prep_kernel<<<(int)((nthreads + 255) / 256), 256, 0, stream>>>(
        x, msg, counts, xbf, xmbf, n_elems);

    // 4) fused dual-GEMM + bias + passthrough
    dim3 grid(D_FEAT / 64, (n_nodes + 127) / 128);
    fused_gemm_kernel<<<grid, 256, 0, stream>>>(xbf, xmbf, x, counts,
                                                W1, b1, W2, b2, out, n_nodes);
}
